// MPNNLayer_38946763441059
// MI455X (gfx1250) — matrix operand order repaired and checked by reading, not compiled, not run on hardware
//
#include <hip/hip_runtime.h>

// ---------------------------------------------------------------------------
// MPNN / CGCNN conv layer for MI455X (gfx1250, wave32, WMMA bf16->f32)
// N=50000 nodes, M=32 neighbors, F=64, NBR=64, H=128, DIN=2F+NBR=192
// ---------------------------------------------------------------------------

typedef __bf16 bf16;
typedef bf16  v16bf __attribute__((ext_vector_type(16)));
typedef bf16  v8bf  __attribute__((ext_vector_type(8)));
typedef float v8f   __attribute__((ext_vector_type(8)));

#define NTOT 50000
#define MNBR 32
#define FF   64
#define HH   128
#define DIN  192
#define NPB  16          // nodes per block (50000 = 3125 * 16)
#define BN_EPS 1e-5f

// ---- dynamic LDS layout (bytes), all chunks 32B aligned ----
#define OFF_W1T 0                          // bf16 [HH][DIN]  = 49152 B
#define OFF_W2T (OFF_W1T + HH*DIN*2)       // bf16 [FF][HH]   = 16384 B
#define OFF_X   (OFF_W2T + FF*HH*2)        // bf16 [MNBR][DIN]= 12288 B
#define OFF_H   (OFF_X + MNBR*DIN*2)       // bf16 [MNBR][HH] =  8192 B
#define OFF_B1  (OFF_H + MNBR*HH*2)        // f32  [HH]
#define OFF_B2  (OFF_B1 + HH*4)            // f32  [FF]
#define OFF_SRC (OFF_B2 + FF*4)            // f32  [FF]
#define OFF_MSK (OFF_SRC + FF*4)           // f32  [MNBR]
#define OFF_IDX (OFF_MSK + MNBR*4)         // i32  [MNBR]
#define OFF_SUM (OFF_IDX + MNBR*4)         // f32  [FF]
#define SMEM_BYTES (OFF_SUM + FF*4)        // = 87552 B (< 320KB/WGP)

__device__ __forceinline__ bf16 f2bf(float f) {
  unsigned u = __builtin_bit_cast(unsigned, f);
  unsigned r = u + 0x7FFFu + ((u >> 16) & 1u);   // round-to-nearest-even
  unsigned short s = (unsigned short)(r >> 16);
  return __builtin_bit_cast(bf16, s);
}

// Branchless, numerically stable softplus:
//   softplus(x) = max(x,0) + log(1 + exp(-|x|))
// Lowers to v_exp_f32 / v_log_f32 straight-line VALU, no EXEC branching.
__device__ __forceinline__ float softplus_f(float v) {
  return fmaxf(v, 0.0f) + __logf(1.0f + __expf(-fabsf(v)));
}

// A fragment (16x32 bf16): lane holds row M = lane%16; K-segments of 8 at
// K = (lane/16)*8 and K = 16 + (lane/16)*8 (ISA 16-bit A layout).
__device__ __forceinline__ v16bf load_afrag(const bf16* rowp, int segA) {
  v8bf lo = *(const v8bf*)(rowp + segA);
  v8bf hi = *(const v8bf*)(rowp + 16 + segA);
  v16bf a;
#pragma unroll
  for (int i = 0; i < 8; ++i) { a[i] = lo[i]; a[8 + i] = hi[i]; }
  return a;
}

__device__ __forceinline__ v8f wmma_bf16(v16bf a, v16bf b, v8f c) {
  return __builtin_amdgcn_wmma_f32_16x16x32_bf16(
      /*neg_a=*/false, a, /*neg_b=*/false, b,
      /*c_mod=*/(short)0, c, /*reuse_a=*/false, /*reuse_b=*/false);
}

// ---------------------------------------------------------------------------
// Phase 1: per-node message passing.  out_pre[n][f] = atom[n][f] + sum_m msg
// ---------------------------------------------------------------------------
__global__ __launch_bounds__(256, 2)
void mpnn_msg_kernel(const float* __restrict__ atom,
                     const float* __restrict__ nbrfea,
                     const long long* __restrict__ nbridx,
                     const float* __restrict__ W1,
                     const float* __restrict__ b1,
                     const float* __restrict__ W2,
                     const float* __restrict__ b2,
                     float* __restrict__ out_pre) {
  extern __shared__ __align__(32) char smem[];
  bf16*  sW1t  = (bf16*)(smem + OFF_W1T);   // [h][k]  K-contiguous
  bf16*  sW2t  = (bf16*)(smem + OFF_W2T);   // [f][h]  K-contiguous
  bf16*  sX    = (bf16*)(smem + OFF_X);     // [m][k]
  bf16*  sH    = (bf16*)(smem + OFF_H);     // [m][h]
  float* sB1   = (float*)(smem + OFF_B1);
  float* sB2   = (float*)(smem + OFF_B2);
  float* sSrc  = (float*)(smem + OFF_SRC);
  float* sMask = (float*)(smem + OFF_MSK);
  int*   sIdx  = (int*)(smem + OFF_IDX);
  float* sMsum = (float*)(smem + OFF_SUM);

  const int tid  = threadIdx.x;
  const int lane = tid & 31;
  const int w    = tid >> 5;              // wave id 0..7
  const int l15  = lane & 15;
  const int segA = (lane >> 4) << 3;      // 0 or 8  (A K-segment)
  const int segB = (lane >> 4) << 4;      // 0 or 16 (unused; B loads use segA)

  // ---- load + transpose weights into LDS (fp32 -> bf16), once per block ----
  for (int i = tid; i < DIN * HH; i += 256) {
    int k = i >> 7, h = i & 127;          // W1 is [DIN][HH] row-major
    sW1t[h * DIN + k] = f2bf(W1[i]);
  }
  for (int i = tid; i < HH * FF; i += 256) {
    int h = i >> 6, f = i & 63;           // W2 is [HH][FF] row-major
    sW2t[f * HH + h] = f2bf(W2[i]);
  }
  if (tid < HH) sB1[tid] = b1[tid];
  if (tid < FF) sB2[tid] = b2[tid];
  __syncthreads();

  for (int ni = 0; ni < NPB; ++ni) {
    const int n = blockIdx.x * NPB + ni;
    if (n >= NTOT) break;                 // block-uniform

    if (tid < MNBR) {
      long long j = nbridx[(long long)n * MNBR + tid];
      sIdx[tid]  = (int)j;
      sMask[tid] = (j == 0) ? 0.0f : 1.0f;
    }
    if (tid < FF) {
      sSrc[tid]  = atom[(long long)n * FF + tid];
      sMsum[tid] = 0.0f;
    }
    __syncthreads();

    // ---- assemble X = [src | gathered nbr | edge] in bf16 ----
    for (int i = tid; i < MNBR * FF; i += 256) {
      int m = i >> 6, f = i & 63;
      sX[m * DIN + f]      = f2bf(sSrc[f]);
      sX[m * DIN + FF + f] = f2bf(atom[(long long)sIdx[m] * FF + f]);
    }
    for (int i = tid; i < MNBR * FF; i += 256) {
      int m = i >> 6, f = i & 63;
      sX[m * DIN + 2 * FF + f] =
          f2bf(nbrfea[((long long)n * MNBR + m) * FF + f]);
    }
    if (ni + 1 < NPB) {   // pull next node's edge rows toward L2
      __builtin_prefetch(nbrfea + ((long long)(n + 1) * MNBR + lane) * FF, 0, 1);
    }
    __syncthreads();

    // ---- GEMM1: h = softplus(X[32x192] * W1[192x128] + b1) ----
    // wave w owns column tile nt=w (cols w*16..), both row tiles.
    v8f acc0 = {0.f, 0.f, 0.f, 0.f, 0.f, 0.f, 0.f, 0.f};
    v8f acc1 = {0.f, 0.f, 0.f, 0.f, 0.f, 0.f, 0.f, 0.f};
    {
      const int colB = w * 16 + l15;      // B lane = output column
#pragma unroll
      for (int kk = 0; kk < DIN / 32; ++kk) {
        v16bf bf = load_afrag(&sW1t[colB * DIN + kk * 32], segA);
        v16bf a0 = load_afrag(&sX[(l15)      * DIN + kk * 32], segA);
        v16bf a1 = load_afrag(&sX[(l15 + 16) * DIN + kk * 32], segA);
        acc0 = wmma_bf16(a0, bf, acc0);
        acc1 = wmma_bf16(a1, bf, acc1);
      }
      // epilogue: bias + softplus, store bf16 H. C layout: M = r + 8*(lane/16)
      const int colN = w * 16 + l15;
      const float bb = sB1[colN];
#pragma unroll
      for (int r = 0; r < 8; ++r) {
        sH[(segA + r) * HH + colN]      = f2bf(softplus_f(acc0[r] + bb));
        sH[(16 + segA + r) * HH + colN] = f2bf(softplus_f(acc1[r] + bb));
      }
    }
    __syncthreads();

    // ---- GEMM2: msg = H[32x128] * W2[128x64] + b2 ; mask ; reduce over m ----
    {
      const int mt2 = w >> 2, nt2 = w & 3;          // 2x4 output tiles
      const int colN2 = nt2 * 16 + l15;
      const int rowA2 = mt2 * 16 + l15;
      v8f acc2 = {0.f, 0.f, 0.f, 0.f, 0.f, 0.f, 0.f, 0.f};
#pragma unroll
      for (int kk = 0; kk < HH / 32; ++kk) {
        v16bf bf = load_afrag(&sW2t[colN2 * HH + kk * 32], segA);
        v16bf af = load_afrag(&sH[rowA2 * HH + kk * 32], segA);
        acc2 = wmma_bf16(af, bf, acc2);
      }
      const float bb = sB2[colN2];
      const int mbase = mt2 * 16 + segA;            // row of acc2[0]
      float p = 0.0f;
#pragma unroll
      for (int r = 0; r < 8; ++r) {
        p += (acc2[r] + bb) * sMask[mbase + r];     // zero masked neighbors
      }
      atomicAdd(&sMsum[colN2], p);                  // ds_add_f32
    }
    __syncthreads();

    if (tid < FF) {
      out_pre[(long long)n * FF + tid] = sSrc[tid] + sMsum[tid];
    }
    __syncthreads();   // protect sSrc/sMsum/sIdx/sX/sH reuse
  }
}

// ---------------------------------------------------------------------------
// Phase 2: batch statistics.  stats[0:64]=sum_f, stats[64:128]=sumsq_f
// ---------------------------------------------------------------------------
__global__ void zero_stats_kernel(float* __restrict__ stats) {
  if (threadIdx.x < 128) stats[threadIdx.x] = 0.0f;
}

__global__ __launch_bounds__(256)
void bn_stats_kernel(const float* __restrict__ pre, float* __restrict__ stats) {
  __shared__ float ssum[256];
  __shared__ float ssq[256];
  const int tid = threadIdx.x;
  const int f   = tid & 63;
  const int grp = tid >> 6;       // 4 row-groups per block
  float s = 0.0f, q = 0.0f;
  for (int r = blockIdx.x * 4 + grp; r < NTOT; r += gridDim.x * 4) {
    float v = pre[(long long)r * FF + f];
    s += v;
    q += v * v;
  }
  ssum[tid] = s;
  ssq[tid]  = q;
  __syncthreads();
  if (tid < 64) {
    float S = ssum[tid] + ssum[tid + 64] + ssum[tid + 128] + ssum[tid + 192];
    float Q = ssq[tid]  + ssq[tid + 64]  + ssq[tid + 128]  + ssq[tid + 192];
    atomicAdd(&stats[tid], S);
    atomicAdd(&stats[64 + tid], Q);
  }
}

// ---------------------------------------------------------------------------
// Phase 3: in-place BatchNorm (batch stats, biased var) + softplus
// ---------------------------------------------------------------------------
__global__ __launch_bounds__(256)
void bn_apply_kernel(float* __restrict__ out, const float* __restrict__ stats,
                     const float* __restrict__ gamma,
                     const float* __restrict__ beta) {
  long long i = (long long)blockIdx.x * 256 + threadIdx.x;
  if (i >= (long long)NTOT * FF) return;
  int f = (int)(i & 63);
  const float inv = 1.0f / (float)NTOT;
  float mu  = stats[f] * inv;
  float var = stats[64 + f] * inv - mu * mu;
  float v = (out[i] - mu) * rsqrtf(var + BN_EPS) * gamma[f] + beta[f];
  out[i] = fmaxf(v, 0.0f) + __logf(1.0f + __expf(-fabsf(v)));
}

// ---------------------------------------------------------------------------
extern "C" void kernel_launch(void* const* d_in, const int* in_sizes, int n_in,
                              void* d_out, int out_size, void* d_ws,
                              size_t ws_size, hipStream_t stream) {
  (void)in_sizes; (void)n_in; (void)out_size; (void)ws_size;
  const float*     atom   = (const float*)d_in[0];
  const float*     nbrfea = (const float*)d_in[1];
  const long long* nbridx = (const long long*)d_in[2];
  const float*     W1     = (const float*)d_in[3];
  const float*     b1     = (const float*)d_in[4];
  const float*     W2     = (const float*)d_in[5];
  const float*     b2     = (const float*)d_in[6];
  const float*     gamma  = (const float*)d_in[7];
  const float*     beta   = (const float*)d_in[8];
  float* out   = (float*)d_out;
  float* stats = (float*)d_ws;    // 128 floats

  const int nblocks = (NTOT + NPB - 1) / NPB;   // 3125
  mpnn_msg_kernel<<<nblocks, 256, SMEM_BYTES, stream>>>(
      atom, nbrfea, nbridx, W1, b1, W2, b2, out);
  zero_stats_kernel<<<1, 128, 0, stream>>>(stats);
  bn_stats_kernel<<<160, 256, 0, stream>>>(out, stats);
  bn_apply_kernel<<<(NTOT * FF + 255) / 256, 256, 0, stream>>>(
      out, stats, gamma, beta);
}
